// SwinTransformerBlock_79637283602561
// MI455X (gfx1250) — compile-verified, matrix operand order checked by reading, not run on hardware
//
#include <hip/hip_runtime.h>
#include <hip/hip_bf16.h>
#include <stdint.h>

typedef __attribute__((ext_vector_type(16))) __bf16 v16bf;
typedef __attribute__((ext_vector_type(8)))  float  v8f;

#define C_DIM   192
#define N_HEADS 6
#define HDIM    32
#define WSZ     8
#define NTOK    64
#define SHIFT_A 4
#define HIMG    128
#define WIMG    128
#define BATCH   8
#define NWIN_TOT (BATCH*256)      // 2048 windows
#define NPIX    (HIMG*WIMG)       // 16384
#define NTOKENS (BATCH*NPIX)      // 131072
#define HID     768

// ---- workspace layout (bytes) ----
#define O_WQKV  ((size_t)0)
#define O_WPROJ (O_WQKV + (size_t)3*C_DIM*C_DIM*2)
#define O_WFC1  (O_WPROJ + (size_t)C_DIM*C_DIM*2)
#define O_WFC2  (O_WFC1 + (size_t)HID*C_DIM*2)
#define O_ATTN  (O_WFC2 + (size_t)C_DIM*HID*2)
#define O_X1    (O_ATTN + (size_t)NWIN_TOT*NTOK*C_DIM*2)
#define O_XN    (O_X1 + (size_t)NTOKENS*C_DIM*4)

// ---------------------------------------------------------------------------
// CDNA5 async memory->LDS copy (GLOBAL_LOAD_ASYNC_TO_LDS_B128, ASYNCcnt).
// Generic pointers to __shared__ carry the LDS byte offset in addr[31:0]
// (ISA 10.2 aperture rule), so we can feed that directly as the VDST operand.
// ---------------------------------------------------------------------------
static __device__ __forceinline__ void async_ld_b128(void* lds_dst, const void* gsrc) {
  uint32_t loff = (uint32_t)(uintptr_t)lds_dst;
  uint64_t ga = (uint64_t)(uintptr_t)gsrc;
  asm volatile("global_load_async_to_lds_b128 %0, %1, off"
               :: "v"(loff), "v"(ga) : "memory");
}
static __device__ __forceinline__ void wait_async() {
  asm volatile("s_wait_asynccnt 0" ::: "memory");
}

// ---------------------------------------------------------------------------
// WMMA helpers (CDNA5, wave32). Layouts per CDNA5 ISA §7.12.2:
//  A (16x32 bf16): lanes 0-15 -> M=lane, K={0..7,16..23}; lanes 16-31 -> K={8..15,24..31}
//  B mirrored with N on the lane axis. C/D: 8 VGPRs, col = lane%16, row = r + 8*(lane/16).
// ---------------------------------------------------------------------------
static __device__ __forceinline__ v8f wmma_bf16(v16bf a, v16bf b, v8f c) {
  return __builtin_amdgcn_wmma_f32_16x16x32_bf16(false, a, false, b, (short)0, c,
                                                 false, false);
}

// Fragment from a row-major [R][K] bf16 array: rows = M (for A) or N (for B=W^T,
// since W is stored [out][in] row-major).
static __device__ __forceinline__ v16bf frag_rm(const __bf16* p, int r0, int k0, int ld) {
  int lane = threadIdx.x & 31;
  const __bf16* base = p + (size_t)(r0 + (lane & 15)) * ld + k0 + ((lane >> 4) * 8);
  v16bf f;
#pragma unroll
  for (int i = 0; i < 8; ++i) { f[i] = base[i]; f[i + 8] = base[16 + i]; }
  return f;
}

// B fragment from a row-major [K][N] bf16 array (e.g. V: keys x dims).
static __device__ __forceinline__ v16bf frag_cm(const __bf16* p, int n0, int k0, int ld) {
  int lane = threadIdx.x & 31;
  int col = n0 + (lane & 15);
  int kh = (lane >> 4) * 8;
  v16bf f;
#pragma unroll
  for (int i = 0; i < 8; ++i) {
    f[i]     = p[(size_t)(k0 + kh + i) * ld + col];
    f[i + 8] = p[(size_t)(k0 + 16 + kh + i) * ld + col];
  }
  return f;
}

// ---------------------------------------------------------------------------
// K0: fp32 -> bf16 weight conversion
// ---------------------------------------------------------------------------
__global__ __launch_bounds__(256) void k_convert(
    const float* __restrict__ qkv_w, const float* __restrict__ proj_w,
    const float* __restrict__ fc1_w, const float* __restrict__ fc2_w,
    __bf16* __restrict__ wq, __bf16* __restrict__ wp,
    __bf16* __restrict__ w1, __bf16* __restrict__ w2) {
  int i = blockIdx.x * 256 + threadIdx.x;
  const int nq = 3 * C_DIM * C_DIM, np = C_DIM * C_DIM;
  const int n1 = HID * C_DIM, n2 = C_DIM * HID;
  if (i < nq) { wq[i] = (__bf16)qkv_w[i]; return; }
  i -= nq;
  if (i < np) { wp[i] = (__bf16)proj_w[i]; return; }
  i -= np;
  if (i < n1) { w1[i] = (__bf16)fc1_w[i]; return; }
  i -= n1;
  if (i < n2) { w2[i] = (__bf16)fc2_w[i]; }
}

// ---------------------------------------------------------------------------
// K1: per-window fused QKV + attention (softmax + rel-bias) -> attn_out (bf16)
// One block = one shifted window (64 tokens x 192 ch), 8 waves.
// ---------------------------------------------------------------------------
__global__ __launch_bounds__(256) void k_attn(
    const float* __restrict__ x, const __bf16* __restrict__ wqkv,
    const float* __restrict__ qkv_b, const float* __restrict__ rel_bias,
    __bf16* __restrict__ attn_out) {
  __shared__ __bf16 sXW[NTOK * C_DIM];   // 24576 B
  __shared__ __bf16 sQ[NTOK * HDIM];     //  4096 B
  __shared__ __bf16 sK[NTOK * HDIM];
  __shared__ __bf16 sV[NTOK * HDIM];
  __shared__ float  sS[NTOK * NTOK];     // 16384 B (reused as bf16 P)

  const int wid = blockIdx.x;
  const int b  = wid >> 8;
  const int wy = (wid >> 4) & 15;
  const int wx = wid & 15;
  const int tid = threadIdx.x;
  const int wave = tid >> 5;
  const int lane = tid & 31;

  // roll(-4,-4) + window partition, fp32 -> bf16 (gather; stays on vmem path)
  for (int e = tid; e < NTOK * C_DIM; e += 256) {
    int n = e & 63, c = e >> 6;
    int sh = (wy * WSZ + (n >> 3) + SHIFT_A) & (HIMG - 1);
    int sw = (wx * WSZ + (n & 7) + SHIFT_A) & (WIMG - 1);
    sXW[n * C_DIM + c] =
        (__bf16)x[(((size_t)b * C_DIM + c) * HIMG + sh) * WIMG + sw];
  }
  __syncthreads();

  for (int h = 0; h < N_HEADS; ++h) {
    // --- QKV: 3 matrices x (4 Mtiles x 2 Ntiles) = 24 tiles, 3 per wave ---
    const int mtq = wave >> 1, ntq = wave & 1;
    for (int which = 0; which < 3; ++which) {
      const int rowbase = which * C_DIM + h * HDIM + ntq * 16;
      v8f acc = {};
#pragma unroll
      for (int kc = 0; kc < 6; ++kc) {
        v16bf a  = frag_rm(sXW, mtq * 16, kc * 32, C_DIM);
        v16bf bf = frag_rm(wqkv, rowbase, kc * 32, C_DIM);
        acc = wmma_bf16(a, bf, acc);
      }
      const float bias = qkv_b[rowbase + (lane & 15)];
      const float scl = (which == 0) ? 0.17677669529663687f : 1.0f;  // 1/sqrt(32) folded into Q
      __bf16* dst = (which == 0) ? sQ : (which == 1) ? sK : sV;
      const int col = ntq * 16 + (lane & 15);
      const int r0 = mtq * 16 + 8 * (lane >> 4);
#pragma unroll
      for (int r = 0; r < 8; ++r)
        dst[(r0 + r) * HDIM + col] = (__bf16)((acc[r] + bias) * scl);
    }
    __syncthreads();

    // --- S = Q K^T + relative position bias : 16 tiles, 2 per wave ---
    for (int j = 0; j < 2; ++j) {
      const int t = wave + 8 * j;
      const int mt = t >> 2, ntb = t & 3;
      v8f acc = {};
      acc = wmma_bf16(frag_rm(sQ, mt * 16, 0, HDIM),
                      frag_rm(sK, ntb * 16, 0, HDIM), acc);
      const int col = ntb * 16 + (lane & 15);
      const int iyk = col >> 3, ixk = col & 7;
      const int r0 = mt * 16 + 8 * (lane >> 4);
#pragma unroll
      for (int r = 0; r < 8; ++r) {
        const int m = r0 + r;
        const int idx = ((m >> 3) - iyk + 7) * 15 + ((m & 7) - ixk + 7);
        sS[m * NTOK + col] = acc[r] + rel_bias[idx * N_HEADS + h];
      }
    }
    __syncthreads();

    // --- softmax over keys; write P (bf16) in place over S ---
    {
      const int r = tid >> 2, q4 = tid & 3;   // 4 lanes per row
      float v[16];
      float mx = -3.4e38f;
#pragma unroll
      for (int i = 0; i < 16; ++i) {
        v[i] = sS[r * NTOK + q4 * 16 + i];
        mx = fmaxf(mx, v[i]);
      }
      mx = fmaxf(mx, __shfl_xor(mx, 1));
      mx = fmaxf(mx, __shfl_xor(mx, 2));
      float sum = 0.f;
#pragma unroll
      for (int i = 0; i < 16; ++i) { v[i] = __expf(v[i] - mx); sum += v[i]; }
      sum += __shfl_xor(sum, 1);
      sum += __shfl_xor(sum, 2);
      const float inv = 1.0f / sum;
      __syncthreads();                         // all f32 reads done
      __bf16* sP = (__bf16*)sS;
#pragma unroll
      for (int i = 0; i < 16; ++i)
        sP[r * NTOK + q4 * 16 + i] = (__bf16)(v[i] * inv);
    }
    __syncthreads();

    // --- O = P V : 8 tiles, 1 per wave ---
    {
      const __bf16* sP = (const __bf16*)sS;
      const int mt = wave >> 1, nt = wave & 1;
      v8f acc = {};
#pragma unroll
      for (int kc = 0; kc < 2; ++kc)
        acc = wmma_bf16(frag_rm(sP, mt * 16, kc * 32, NTOK),
                        frag_cm(sV, nt * 16, kc * 32, HDIM), acc);
      const int col = nt * 16 + (lane & 15);
      const int r0 = mt * 16 + 8 * (lane >> 4);
      __bf16* dst = attn_out + (size_t)wid * NTOK * C_DIM + h * HDIM + col;
#pragma unroll
      for (int r = 0; r < 8; ++r)
        dst[(size_t)(r0 + r) * C_DIM] = (__bf16)acc[r];
    }
    __syncthreads();
  }
}

// ---------------------------------------------------------------------------
// K2: per-window output projection + (row-major-reshape) window reverse + roll
// Tile staging uses CDNA5 async memory->LDS (no VGPR round-trip).
// ---------------------------------------------------------------------------
__global__ __launch_bounds__(256) void k_proj(
    const __bf16* __restrict__ attn_out, const __bf16* __restrict__ wproj,
    const float* __restrict__ proj_b, float* __restrict__ x1) {
  __shared__ __bf16 sA[NTOK * C_DIM];
  const int wid = blockIdx.x, tid = threadIdx.x;
  const int wave = tid >> 5, lane = tid & 31;
  {
    const __bf16* src = attn_out + (size_t)wid * NTOK * C_DIM;
    for (int e = tid; e < NTOK * C_DIM / 8; e += 256)      // 1536 x 16B
      async_ld_b128(sA + e * 8, src + e * 8);
    wait_async();
  }
  __syncthreads();
  const int bimg = wid >> 8, wy = (wid >> 4) & 15, wx = wid & 15;
  for (int j = 0; j < 6; ++j) {
    const int t = wave + 8 * j;
    const int mt = t & 3, ntile = t >> 2;
    v8f acc = {};
#pragma unroll
    for (int kc = 0; kc < 6; ++kc)
      acc = wmma_bf16(frag_rm(sA, mt * 16, kc * 32, C_DIM),
                      frag_rm(wproj, ntile * 16, kc * 32, C_DIM), acc);
    const int col = ntile * 16 + (lane & 15);
    const float bias = proj_b[col];
    const int r0 = mt * 16 + 8 * (lane >> 4);
#pragma unroll
    for (int r = 0; r < 8; ++r) {
      const int m = r0 + r;
      // faithful torch-view reshape: flat g reinterpreted as (C,H,W)
      const int g = (((wy * 8 + (m >> 3)) * 16 + wx) * 8 + (m & 7)) * C_DIM + col;
      const int c2 = g >> 14;
      const int rem = g & 16383;
      const int hh = ((rem >> 7) + SHIFT_A) & 127;
      const int ww = ((rem & 127) + SHIFT_A) & 127;
      x1[(((size_t)bimg * C_DIM + c2) * HIMG + hh) * WIMG + ww] = acc[r] + bias;
    }
  }
}

// ---------------------------------------------------------------------------
// K3: x2 = x + LN1(x1)  (-> d_out) ;  xn = LN2(x2) (bf16, token-major)
// Thread-per-pixel: consecutive threads hit consecutive pixels (coalesced).
// ---------------------------------------------------------------------------
__global__ __launch_bounds__(256) void k_ln(
    const float* __restrict__ x, const float* __restrict__ x1,
    const float* __restrict__ g1, const float* __restrict__ b1,
    const float* __restrict__ g2, const float* __restrict__ b2,
    float* __restrict__ out, __bf16* __restrict__ xn) {
  const int p = blockIdx.x * 256 + threadIdx.x;
  const int b = p >> 14;
  const int pix = p & 16383;
  const float* px1 = x1 + (size_t)b * C_DIM * NPIX + pix;
  const float* px  = x  + (size_t)b * C_DIM * NPIX + pix;
  float* po = out + (size_t)b * C_DIM * NPIX + pix;

  float s = 0.f, ss = 0.f;
  for (int c = 0; c < C_DIM; ++c) {
    const float v = px1[(size_t)c * NPIX];
    s += v; ss += v * v;
  }
  const float mu = s * (1.0f / C_DIM);
  const float rs = rsqrtf(ss * (1.0f / C_DIM) - mu * mu + 1e-5f);

  float s2 = 0.f, ss2 = 0.f;
  for (int c = 0; c < C_DIM; ++c) {
    const float v = (px1[(size_t)c * NPIX] - mu) * rs * g1[c] + b1[c] +
                    px[(size_t)c * NPIX];
    po[(size_t)c * NPIX] = v;
    s2 += v; ss2 += v * v;
  }
  const float mu2 = s2 * (1.0f / C_DIM);
  const float rs2 = rsqrtf(ss2 * (1.0f / C_DIM) - mu2 * mu2 + 1e-5f);
  __bf16* pxn = xn + (size_t)p * C_DIM;
  for (int c = 0; c < C_DIM; ++c) {
    const float v = po[(size_t)c * NPIX];
    pxn[c] = (__bf16)((v - mu2) * rs2 * g2[c] + b2[c]);
  }
}

// ---------------------------------------------------------------------------
// K4: MLP  d_out += fc2(gelu(fc1(xn))) ; 64-token tiles, hidden in 128-chunks
// ---------------------------------------------------------------------------
__global__ __launch_bounds__(256) void k_mlp(
    const __bf16* __restrict__ xn, const __bf16* __restrict__ w1,
    const float* __restrict__ b1v, const __bf16* __restrict__ w2,
    const float* __restrict__ b2v, float* __restrict__ out) {
  __shared__ __bf16 sA[NTOK * C_DIM];   // 24576 B
  __shared__ __bf16 sH[NTOK * 128];     // 16384 B
  const int tid = threadIdx.x, wave = tid >> 5, lane = tid & 31;
  const size_t tokbase = (size_t)blockIdx.x * NTOK;
  {
    const __bf16* src = xn + tokbase * C_DIM;
    for (int e = tid; e < NTOK * C_DIM / 8; e += 256)      // async tile stage
      async_ld_b128(sA + e * 8, src + e * 8);
    wait_async();
  }
  __syncthreads();

  v8f acc[6] = {};
  for (int cc = 0; cc < 6; ++cc) {
    if (cc + 1 < 6)  // pull next fc1 weight chunk into cache (global_prefetch)
      __builtin_prefetch(w1 + (size_t)(cc + 1) * 128 * C_DIM, 0, 3);
    // fc1 chunk of 128 hidden units: 32 tiles, 4 per wave
#pragma unroll
    for (int i = 0; i < 4; ++i) {
      const int t = wave + 8 * i;
      const int mt = t & 3, nh = t >> 2;
      v8f a1 = {};
#pragma unroll
      for (int kc = 0; kc < 6; ++kc)
        a1 = wmma_bf16(frag_rm(sA, mt * 16, kc * 32, C_DIM),
                       frag_rm(w1, cc * 128 + nh * 16, kc * 32, C_DIM), a1);
      const int col = nh * 16 + (lane & 15);
      const float bias = b1v[cc * 128 + col];
      const int r0 = mt * 16 + 8 * (lane >> 4);
#pragma unroll
      for (int r = 0; r < 8; ++r) {
        const float xv = a1[r] + bias;
        const float ge = 0.5f * xv * (1.0f + erff(xv * 0.7071067811865475f));
        sH[(r0 + r) * 128 + col] = (__bf16)ge;
      }
    }
    __syncthreads();
    // fc2 partial accumulation over this hidden chunk
#pragma unroll
    for (int j = 0; j < 6; ++j) {
      const int t = wave + 8 * j;
      const int mt = t & 3, ntile = t >> 2;
#pragma unroll
      for (int kc = 0; kc < 4; ++kc)
        acc[j] = wmma_bf16(frag_rm(sH, mt * 16, kc * 32, 128),
                           frag_rm(w2, ntile * 16, cc * 128 + kc * 32, HID),
                           acc[j]);
    }
    __syncthreads();
  }
  // epilogue: scatter back to (B,C,H,W) and accumulate the residual
#pragma unroll
  for (int j = 0; j < 6; ++j) {
    const int t = wave + 8 * j;
    const int mt = t & 3, ntile = t >> 2;
    const int col = ntile * 16 + (lane & 15);
    const float bias = b2v[col];
    const int r0 = mt * 16 + 8 * (lane >> 4);
#pragma unroll
    for (int r = 0; r < 8; ++r) {
      const size_t tok = tokbase + r0 + r;
      const size_t idx = ((tok >> 14) * C_DIM + col) * NPIX + (tok & 16383);
      out[idx] += acc[j][r] + bias;
    }
  }
}

// ---------------------------------------------------------------------------
extern "C" void kernel_launch(void* const* d_in, const int* in_sizes, int n_in,
                              void* d_out, int out_size, void* d_ws, size_t ws_size,
                              hipStream_t stream) {
  const float* x      = (const float*)d_in[0];
  const float* qkv_w  = (const float*)d_in[1];
  const float* qkv_b  = (const float*)d_in[2];
  const float* proj_w = (const float*)d_in[3];
  const float* proj_b = (const float*)d_in[4];
  const float* rel_b  = (const float*)d_in[5];
  const float* n1g    = (const float*)d_in[6];
  const float* n1b    = (const float*)d_in[7];
  const float* n2g    = (const float*)d_in[8];
  const float* n2b    = (const float*)d_in[9];
  const float* fc1_w  = (const float*)d_in[10];
  const float* fc1_b  = (const float*)d_in[11];
  const float* fc2_w  = (const float*)d_in[12];
  const float* fc2_b  = (const float*)d_in[13];

  char* ws = (char*)d_ws;
  __bf16* wq   = (__bf16*)(ws + O_WQKV);
  __bf16* wp   = (__bf16*)(ws + O_WPROJ);
  __bf16* w1   = (__bf16*)(ws + O_WFC1);
  __bf16* w2   = (__bf16*)(ws + O_WFC2);
  __bf16* attn = (__bf16*)(ws + O_ATTN);
  float*  x1   = (float*)(ws + O_X1);
  __bf16* xnb  = (__bf16*)(ws + O_XN);
  float*  out  = (float*)d_out;

  const int n_wconv = 3*C_DIM*C_DIM + C_DIM*C_DIM + HID*C_DIM + C_DIM*HID; // 442368
  k_convert<<<(n_wconv + 255) / 256, 256, 0, stream>>>(qkv_w, proj_w, fc1_w, fc2_w,
                                                       wq, wp, w1, w2);
  k_attn<<<NWIN_TOT, 256, 0, stream>>>(x, wq, qkv_b, rel_b, attn);
  k_proj<<<NWIN_TOT, 256, 0, stream>>>(attn, wp, proj_b, x1);
  k_ln<<<NTOKENS / 256, 256, 0, stream>>>(x, x1, n1g, n1b, n2g, n2b, out, xnb);
  k_mlp<<<NTOKENS / NTOK, 256, 0, stream>>>(xnb, w1, fc1_b, w2, fc2_b, out);
}
